// TransformerDecoderAttnLayer_35837207118651
// MI455X (gfx1250) — matrix operand order repaired and checked by reading, not compiled, not run on hardware
//
#include <hip/hip_runtime.h>
#include <hip/hip_bf16.h>

#define B_ 2
#define S_ 2048
#define HIDDEN_ 4096
#define HD_ 128
#define HQ_ 32
#define HKV_ 8
#define WINDOW_ 1024
#define CAP_ 50.0f
#define EPS_ 1e-5f
#define NEG_INF_ (-1e30f)
#define ROPE_BASE_ 10000.0f

typedef unsigned int   u32;
typedef unsigned short u16;
typedef __attribute__((ext_vector_type(16))) __bf16 v16bf;
typedef __attribute__((ext_vector_type(8)))  float  v8f;

union FragB16 { v16bf v; u32 u[8]; };

__device__ __forceinline__ float bf2f(u16 h) { u32 x = ((u32)h) << 16; return __uint_as_float(x); }
__device__ __forceinline__ u16 f2bf(float f) {
  u32 x = __float_as_uint(f);
  x = x + 0x7FFFu + ((x >> 16) & 1u);   // round-to-nearest-even
  return (u16)(x >> 16);
}

// ---------------------------------------------------------------- RMSNorm -> bf16
__global__ __launch_bounds__(256) void rmsnorm_kernel(const float* __restrict__ x,
                                                      const float* __restrict__ gamma,
                                                      u16* __restrict__ xn) {
  const int row = blockIdx.x;            // 0 .. B*S-1
  const int tid = threadIdx.x;
  const float* xr = x + (size_t)row * HIDDEN_;
  float vals[16];
  float ss = 0.f;
#pragma unroll
  for (int i = 0; i < 16; ++i) { float v = xr[tid * 16 + i]; vals[i] = v; ss += v * v; }
  __shared__ float red[256];
  red[tid] = ss; __syncthreads();
  for (int s = 128; s > 0; s >>= 1) { if (tid < s) red[tid] += red[tid + s]; __syncthreads(); }
  const float scale = rsqrtf(red[0] / (float)HIDDEN_ + EPS_);
  u16* outr = xn + (size_t)row * HIDDEN_ + tid * 16;
#pragma unroll
  for (int i = 0; i < 16; ++i) outr[i] = f2bf(vals[i] * scale * gamma[tid * 16 + i]);
}

// ---------------------------------------------------------------- f32 [R][C] -> bf16 transposed [C][R]
__global__ __launch_bounds__(256) void transpose_cvt_kernel(const float* __restrict__ src,
                                                            u16* __restrict__ dst,
                                                            int R, int C) {
  __shared__ u16 tile[32][33];
  const int bx = blockIdx.x * 32;              // along C
  const int by = blockIdx.y * 32;              // along R
  const int tx = threadIdx.x & 31, ty = threadIdx.x >> 5;   // 32x8
#pragma unroll
  for (int i = 0; i < 4; ++i)
    tile[ty + i * 8][tx] = f2bf(src[(size_t)(by + ty + i * 8) * C + bx + tx]);
  __syncthreads();
#pragma unroll
  for (int i = 0; i < 4; ++i)
    dst[(size_t)(bx + ty + i * 8) * R + by + tx] = tile[tx][ty + i * 8];
}

// ---------------------------------------------------------------- bf16 WMMA GEMM
// C[M,N] = A[M,K] @ W[K,N] with W pre-transposed: WT[N][K] (bf16 row-major).
// Block tile 64x256, 8 waves (2x4), wave tile 32x64 (2x4 WMMA), K-step 32, double-buffered LDS.
// MODE 0: bf16 store [M][N]; MODE 1: f32 + residual; MODE 2: bf16 transposed V [b][g][d][s].
template <int MODE>
__global__ __launch_bounds__(256) void gemm_bf16_wmma_kernel(
    const u16* __restrict__ A, const u16* __restrict__ WT,
    u16* __restrict__ out_bf, float* __restrict__ out_f, const float* __restrict__ resid,
    int M, int N, int K) {
  __shared__ __align__(16) u16 As[2][64 * 32];    // row-major, stride 32
  __shared__ __align__(16) u16 Bt[2][256 * 32];   // Bt[n][k], stride 32

  const int tid  = threadIdx.x;
  const int wave = tid >> 5, lane = tid & 31;
  const int l15 = lane & 15, kh = lane >> 4;
  const int wr = wave >> 2, wc = wave & 3;
  const int m0 = blockIdx.x * 64, n0 = blockIdx.y * 256;

  const v8f vzero = {0.f, 0.f, 0.f, 0.f, 0.f, 0.f, 0.f, 0.f};
  v8f acc[2][4];
#pragma unroll
  for (int mt = 0; mt < 2; ++mt)
#pragma unroll
    for (int nt = 0; nt < 4; ++nt) acc[mt][nt] = vzero;

  const int arow = tid >> 2, aseg = (tid & 3) * 8;           // A: one uint4/thread
  const u16* aptr = A + (size_t)(m0 + arow) * K + aseg;
  const u16* bptr = WT + (size_t)(n0 + tid) * K;             // B: one 32-elem row/thread

  *(uint4*)&As[0][arow * 32 + aseg] = *(const uint4*)aptr;
#pragma unroll
  for (int j = 0; j < 4; ++j)
    *(uint4*)&Bt[0][tid * 32 + j * 8] = *(const uint4*)(bptr + j * 8);

  int cur = 0;
  for (int k0 = 0; k0 < K; k0 += 32) {
    __syncthreads();
    uint4 an = {};
    uint4 bn[4] = {};
    const bool more = (k0 + 32) < K;
    if (more) {
      an = *(const uint4*)(aptr + k0 + 32);
#pragma unroll
      for (int j = 0; j < 4; ++j) bn[j] = *(const uint4*)(bptr + k0 + 32 + j * 8);
      if (k0 + 64 < K) {
        __builtin_prefetch(aptr + k0 + 64, 0, 0);
        __builtin_prefetch(bptr + k0 + 64, 0, 0);
      }
    }

    FragB16 af[2], bfr[4];
#pragma unroll
    for (int mt = 0; mt < 2; ++mt) {      // A-fragment: row = l15, K pairs per ISA layout
      const int r = wr * 32 + mt * 16 + l15;
#pragma unroll
      for (int v = 0; v < 8; ++v) {
        const int kk = ((v >> 2) << 4) + kh * 8 + (v & 3) * 2;
        af[mt].u[v] = *(const u32*)&As[cur][r * 32 + kk];
      }
    }
#pragma unroll
    for (int nt = 0; nt < 4; ++nt) {      // B-fragment: col = l15, K pairs per ISA layout
      const int c = wc * 64 + nt * 16 + l15;
#pragma unroll
      for (int v = 0; v < 8; ++v)
        bfr[nt].u[v] = *(const u32*)&Bt[cur][c * 32 + ((v >> 2) << 4) + kh * 8 + (v & 3) * 2];
    }
#pragma unroll
    for (int mt = 0; mt < 2; ++mt)
#pragma unroll
      for (int nt = 0; nt < 4; ++nt)
        acc[mt][nt] = __builtin_amdgcn_wmma_f32_16x16x32_bf16(
            false, af[mt].v, false, bfr[nt].v, (short)0, acc[mt][nt], false, false);

    if (more) {
      *(uint4*)&As[cur ^ 1][arow * 32 + aseg] = an;
#pragma unroll
      for (int j = 0; j < 4; ++j) *(uint4*)&Bt[cur ^ 1][tid * 32 + j * 8] = bn[j];
    }
    cur ^= 1;
  }

  // D layout: element e -> row e + 8*kh, col l15.
#pragma unroll
  for (int mt = 0; mt < 2; ++mt)
#pragma unroll
    for (int nt = 0; nt < 4; ++nt) {
      const int n = n0 + wc * 64 + nt * 16 + l15;
#pragma unroll
      for (int e = 0; e < 8; ++e) {
        const int m = m0 + wr * 32 + mt * 16 + e + kh * 8;
        if (MODE == 0) {
          out_bf[(size_t)m * N + n] = f2bf(acc[mt][nt][e]);
        } else if (MODE == 1) {
          const size_t idx = (size_t)m * N + n;
          out_f[idx] = acc[mt][nt][e] + resid[idx];
        } else {
          const size_t idx = ((size_t)(m >> 11) * (HKV_ * HD_) + n) * S_ + (m & (S_ - 1));
          out_bf[idx] = f2bf(acc[mt][nt][e]);
        }
      }
    }
}

// ---------------------------------------------------------------- in-place RoPE on bf16 [B,S,nh,HD]
__global__ __launch_bounds__(256) void rope_kernel(u16* __restrict__ t, int nh, int total) {
  const int idx = blockIdx.x * 256 + threadIdx.x;
  if (idx >= total) return;
  const int j = idx & 63;
  const int h = (idx >> 6) % nh;
  const int s = ((idx >> 6) / nh) % S_;
  const int b = ((idx >> 6) / nh) / S_;
  const float inv = powf(ROPE_BASE_, -(float)j / 64.0f);
  const float ang = (float)s * inv;
  const float c = cosf(ang), sn = sinf(ang);
  const size_t base = (((size_t)b * S_ + s) * nh + h) * HD_;
  const float t1 = bf2f(t[base + j]), t2 = bf2f(t[base + 64 + j]);
  t[base + j]      = f2bf(t1 * c - t2 * sn);
  t[base + 64 + j] = f2bf(t2 * c + t1 * sn);
}

// ---------------------------------------------------------------- flash attention (GQA, window, softcap)
// 256-thread blocks (8 waves = 4 reps x 2 query tiles) so each wave keeps a large VGPR
// budget (2 waves/SIMD) -- no scratch spills around the WMMA loop.
// Grid: B * HKV * (S/32).  K given as [b][s][g][d] (roped), V pre-transposed [b][g][d][s].
__global__ __launch_bounds__(256) void attn_kernel(const u16* __restrict__ Q,
                                                   const u16* __restrict__ Kb,
                                                   const u16* __restrict__ VT,
                                                   u16* __restrict__ O) {
  __shared__ __align__(16) u16 Ks[32 * 128];      // K tile row-major [k][d]
  __shared__ __align__(16) u16 Vt[128 * 32];      // V tile [d][k]
  __shared__ __align__(16) u16 Pb[8 * 16 * 32];   // per-wave prob staging 16x32

  const int tid  = threadIdx.x;
  const int wave = tid >> 5, lane = tid & 31;
  const int l15 = lane & 15, kh = lane >> 4;

  const int bx = blockIdx.x;
  const int qc = bx & 63;            // query chunk (S/32 = 64)
  const int g  = (bx >> 6) & 7;      // kv head
  const int b  = bx >> 9;            // batch

  const int r  = wave >> 1;          // rep 0..3
  const int qt = wave & 1;           // query tile within chunk
  const int h  = g * 4 + r;
  const int qbase = qc * 32 + qt * 16;

  // Q fragment (16x128) split into 4 K-slices of 32
  FragB16 qf[4];
  {
    const size_t qrow = (((size_t)b * S_ + qbase + l15) * HQ_ + h) * HD_;
#pragma unroll
    for (int ds = 0; ds < 4; ++ds)
#pragma unroll
      for (int v = 0; v < 8; ++v) {
        const int kk = ds * 32 + ((v >> 2) << 4) + kh * 8 + (v & 3) * 2;
        qf[ds].u[v] = *(const u32*)&Q[qrow + kk];
      }
  }

  const v8f vzero = {0.f, 0.f, 0.f, 0.f, 0.f, 0.f, 0.f, 0.f};
  v8f Oa[8];
#pragma unroll
  for (int t = 0; t < 8; ++t) Oa[t] = vzero;
  float mrow[8], lrow[8];
#pragma unroll
  for (int e = 0; e < 8; ++e) { mrow[e] = NEG_INF_; lrow[e] = 0.f; }

  int kstart = qc * 32 - (WINDOW_ - 1);
  if (kstart < 0) kstart = 0;
  kstart &= ~31;
  const int kend = qc * 32 + 31;

  const int krow_s = tid >> 3;          // K staging: key row 0..31
  const int kcol_s = (tid & 7) * 16;    // K staging: 16 d's (2 x uint4)
  const int vd = tid >> 1;              // V staging: d 0..127
  const int vk = (tid & 1) * 16;        // V staging: 16 k's (2 x uint4)
  const size_t vtb = (size_t)(b * HKV_ + g) * HD_ * S_;

  for (int k0 = kstart; k0 <= kend; k0 += 32) {
    {
      const size_t kb = (((size_t)b * S_ + (k0 + krow_s)) * HKV_ + g) * HD_ + kcol_s;
      *(uint4*)&Ks[krow_s * 128 + kcol_s]     = *(const uint4*)&Kb[kb];
      *(uint4*)&Ks[krow_s * 128 + kcol_s + 8] = *(const uint4*)&Kb[kb + 8];
      const size_t vb = vtb + (size_t)vd * S_ + k0 + vk;
      *(uint4*)&Vt[vd * 32 + vk]     = *(const uint4*)&VT[vb];
      *(uint4*)&Vt[vd * 32 + vk + 8] = *(const uint4*)&VT[vb + 8];
      if (k0 + 32 <= kend) {
        __builtin_prefetch(&Kb[kb + (size_t)32 * HKV_ * HD_], 0, 0);
        __builtin_prefetch(&VT[vb + 32], 0, 0);
      }
    }
    __syncthreads();

    // scores S = Q·K^T for 2 key tiles (Ks row-major == transposed B operand)
    float sc[2][8];
#pragma unroll
    for (int nt = 0; nt < 2; ++nt) {
      v8f s = vzero;
      const int krow = nt * 16 + l15;
#pragma unroll
      for (int ds = 0; ds < 4; ++ds) {
        FragB16 bfr;
#pragma unroll
        for (int v = 0; v < 8; ++v)
          bfr.u[v] = *(const u32*)&Ks[krow * 128 + ds * 32 + ((v >> 2) << 4) + kh * 8 + (v & 3) * 2];
        s = __builtin_amdgcn_wmma_f32_16x16x32_bf16(false, qf[ds].v, false, bfr.v,
                                                    (short)0, s, false, false);
      }
#pragma unroll
      for (int e = 0; e < 8; ++e) {
        float xv = s[e] * 0.08838834764831845f;   // 1/sqrt(128)
        xv = CAP_ * tanhf(xv * (1.0f / CAP_));
        const int i = qbase + e + kh * 8;
        const int j = k0 + nt * 16 + l15;
        const bool valid = (j <= i) && (j > i - WINDOW_);
        sc[nt][e] = valid ? xv : NEG_INF_;
      }
    }

    // online softmax per row (rows 0-7 on lanes 0-15, rows 8-15 on lanes 16-31);
    // sc[] is overwritten with the probabilities to keep live registers low.
#pragma unroll
    for (int e = 0; e < 8; ++e) {
      float mx = fmaxf(sc[0][e], sc[1][e]);
#pragma unroll
      for (int off = 1; off < 16; off <<= 1) mx = fmaxf(mx, __shfl_xor(mx, off, 32));
      const float mnew = fmaxf(mrow[e], mx);
      const float f = __expf(mrow[e] - mnew);
      const float p0 = __expf(sc[0][e] - mnew);
      const float p1 = __expf(sc[1][e] - mnew);
      float ssum = p0 + p1;
#pragma unroll
      for (int off = 1; off < 16; off <<= 1) ssum += __shfl_xor(ssum, off, 32);
      lrow[e] = lrow[e] * f + ssum;
      mrow[e] = mnew;
#pragma unroll
      for (int t = 0; t < 8; ++t) Oa[t][e] *= f;
      sc[0][e] = p0; sc[1][e] = p1;
    }

    // re-layout probs D->A via per-wave LDS patch (same-wave DS ops are in order)
    u16* Pw = &Pb[wave * 512];
#pragma unroll
    for (int nt = 0; nt < 2; ++nt)
#pragma unroll
      for (int e = 0; e < 8; ++e)
        Pw[(e + kh * 8) * 32 + nt * 16 + l15] = f2bf(sc[nt][e]);
    FragB16 pf;
#pragma unroll
    for (int v = 0; v < 8; ++v) {
      const int kk = ((v >> 2) << 4) + kh * 8 + (v & 3) * 2;
      pf.u[v] = *(const u32*)&Pw[l15 * 32 + kk];
    }

    // O += P · V  (Vt[d][k] == transposed B operand)
#pragma unroll
    for (int t = 0; t < 8; ++t) {
      FragB16 vf;
#pragma unroll
      for (int v = 0; v < 8; ++v)
        vf.u[v] = *(const u32*)&Vt[(t * 16 + l15) * 32 + ((v >> 2) << 4) + kh * 8 + (v & 3) * 2];
      Oa[t] = __builtin_amdgcn_wmma_f32_16x16x32_bf16(false, pf.v, false, vf.v,
                                                      (short)0, Oa[t], false, false);
    }
    __syncthreads();
  }

  // normalize & store bf16 [b][s][h][d]
#pragma unroll
  for (int e = 0; e < 8; ++e) {
    const float inv = 1.0f / lrow[e];
    const int i = qbase + e + kh * 8;
    const size_t base = (((size_t)b * S_ + i) * HQ_ + h) * HD_;
#pragma unroll
    for (int t = 0; t < 8; ++t)
      O[base + t * 16 + l15] = f2bf(Oa[t][e] * inv);
  }
}

// ---------------------------------------------------------------- host-side orchestration
extern "C" void kernel_launch(void* const* d_in, const int* in_sizes, int n_in,
                              void* d_out, int out_size, void* d_ws, size_t ws_size,
                              hipStream_t stream) {
  (void)in_sizes; (void)n_in; (void)out_size; (void)ws_size;
  const float* x     = (const float*)d_in[0];
  const float* gamma = (const float*)d_in[1];
  const float* Wq    = (const float*)d_in[2];
  const float* Wk    = (const float*)d_in[3];
  const float* Wv    = (const float*)d_in[4];
  const float* Wo    = (const float*)d_in[5];
  float* out = (float*)d_out;

  const int M = B_ * S_;               // 4096
  const int NQ = HQ_ * HD_;            // 4096
  const int NKV = HKV_ * HD_;          // 1024

  char* ws = (char*)d_ws;
  size_t off = 0;
  auto carve = [&](size_t elems) -> u16* {
    u16* p = (u16*)(ws + off);
    off += (elems * 2 + 255) & ~(size_t)255;
    return p;
  };
  u16* WqT     = carve((size_t)NQ * HIDDEN_);     // [N][K]
  u16* WkT     = carve((size_t)NKV * HIDDEN_);
  u16* WvT     = carve((size_t)NKV * HIDDEN_);
  u16* WoT     = carve((size_t)HIDDEN_ * NQ);
  u16* xn_bf   = carve((size_t)M * HIDDEN_);
  u16* q_bf    = carve((size_t)M * NQ);
  u16* k_bf    = carve((size_t)M * NKV);
  u16* vT_bf   = carve((size_t)M * NKV);          // [b][g][d][s]
  u16* attn_bf = carve((size_t)M * NQ);

  // 1) RMSNorm
  rmsnorm_kernel<<<M, 256, 0, stream>>>(x, gamma, xn_bf);

  // 2) weight transpose + bf16 conversion (one-time, coalesced both ways)
  transpose_cvt_kernel<<<dim3(NQ / 32, HIDDEN_ / 32), 256, 0, stream>>>(Wq, WqT, HIDDEN_, NQ);
  transpose_cvt_kernel<<<dim3(NKV / 32, HIDDEN_ / 32), 256, 0, stream>>>(Wk, WkT, HIDDEN_, NKV);
  transpose_cvt_kernel<<<dim3(NKV / 32, HIDDEN_ / 32), 256, 0, stream>>>(Wv, WvT, HIDDEN_, NKV);
  transpose_cvt_kernel<<<dim3(HIDDEN_ / 32, NQ / 32), 256, 0, stream>>>(Wo, WoT, NQ, HIDDEN_);

  // 3) QKV projections (bf16 WMMA GEMM)
  gemm_bf16_wmma_kernel<0><<<dim3(M / 64, NQ / 256), 256, 0, stream>>>(
      xn_bf, WqT, q_bf, nullptr, nullptr, M, NQ, HIDDEN_);
  gemm_bf16_wmma_kernel<0><<<dim3(M / 64, NKV / 256), 256, 0, stream>>>(
      xn_bf, WkT, k_bf, nullptr, nullptr, M, NKV, HIDDEN_);
  gemm_bf16_wmma_kernel<2><<<dim3(M / 64, NKV / 256), 256, 0, stream>>>(
      xn_bf, WvT, vT_bf, nullptr, nullptr, M, NKV, HIDDEN_);   // writes V transposed

  // 4) RoPE in place on Q and K
  {
    const int tq = B_ * S_ * HQ_ * 64;
    rope_kernel<<<(tq + 255) / 256, 256, 0, stream>>>(q_bf, HQ_, tq);
    const int tk = B_ * S_ * HKV_ * 64;
    rope_kernel<<<(tk + 255) / 256, 256, 0, stream>>>(k_bf, HKV_, tk);
  }

  // 5) attention (256-thread blocks, 32-query chunks)
  attn_kernel<<<B_ * HKV_ * (S_ / 32), 256, 0, stream>>>(q_bf, k_bf, vT_bf, attn_bf);

  // 6) output projection + residual (f32 out)
  gemm_bf16_wmma_kernel<1><<<dim3(M / 64, HIDDEN_ / 256), 256, 0, stream>>>(
      attn_bf, WoT, nullptr, out, x, M, HIDDEN_, NQ);
}
